// GlobalFeatureExtractor_80582176408290
// MI455X (gfx1250) — matrix operand order repaired and checked by reading, not compiled, not run on hardware
//
#include <hip/hip_runtime.h>
#include <hip/hip_bf16.h>
#include <math.h>

#define B_    16
#define S_    1024
#define CIN_  1280
#define D_    512
#define H_    8
#define HD_   64
#define ROWS_ (B_*S_)   /* 16384 */
#define BH_   (B_*H_)   /* 128   */

typedef __bf16 v16bf __attribute__((ext_vector_type(16)));
typedef float  v8f   __attribute__((ext_vector_type(8)));
typedef int    v4i   __attribute__((ext_vector_type(4)));

union FragBF { v16bf v; uint4 q[2]; };

static __device__ __forceinline__ v8f wmma_bf16(const FragBF& a, const FragBF& b, v8f c) {
  return __builtin_amdgcn_wmma_f32_16x16x32_bf16(false, a.v, false, b.v, (short)0, c, false, false);
}

// --------------------------- async LDS copy path ---------------------------
#if __has_builtin(__builtin_amdgcn_global_load_async_to_lds_b128)
#define ASYNC_LDS 1
#else
#define ASYNC_LDS 0
#endif
#if __has_builtin(__builtin_amdgcn_s_wait_asynccnt)
#define HAVE_WAITASYNC 1
#else
#define HAVE_WAITASYNC 0
#endif

typedef __attribute__((address_space(1))) v4i* gv4i_p;  // global-AS vector ptr
typedef __attribute__((address_space(3))) v4i* lv4i_p;  // LDS-AS vector ptr

static __device__ __forceinline__ void async_copy16(const __bf16* g, __bf16* l) {
#if ASYNC_LDS
  __builtin_amdgcn_global_load_async_to_lds_b128(
      (gv4i_p)(size_t)g,
      (lv4i_p)(unsigned)(size_t)l,
      0, 0);
#else
  *reinterpret_cast<uint4*>(l) = *reinterpret_cast<const uint4*>(g);
#endif
}

static __device__ __forceinline__ void async_wait_all() {
#if ASYNC_LDS
#if HAVE_WAITASYNC
  __builtin_amdgcn_s_wait_asynccnt(0);
#else
  asm volatile("s_wait_asynccnt 0x0" ::: "memory");
#endif
#endif
}

// ---------------------------------------------------------------------------
// Core: one block (4 waves) computes 32(M) x 256(N); each wave 32x64.
// A slab (32 x 64K bf16 = 4KB) staged in LDS via async copies, double-buffered.
// B fragments loaded direct from global (Bt is N x K row-major = B transposed),
// all 16 b128 loads issued up-front per slab so waits overlap the WMMA pack.
// A-frag: lane=row M%16; elems 0..15 = K 16*(lane/16).. contiguous.
// B-frag: lane=col N%16; elems 0..15 = K 16*(lane/16).. contiguous.
// ---------------------------------------------------------------------------
static __device__ __forceinline__ void gemm_core_32x64(const __bf16* __restrict__ A,
                                                       const __bf16* __restrict__ Bt,
                                                       int K, int m0, int n0, int tid,
                                                       __bf16 (*Asl)[32][64],
                                                       v8f acc[2][4]) {
  const int lane = tid & 31;
  const int lm = lane & 15, lg = lane >> 4;

  const __bf16* bp[4];
#pragma unroll
  for (int t = 0; t < 4; ++t)
    bp[t] = Bt + (size_t)(n0 + t * 16 + lm) * K + 16 * lg;

  // slab staging: 256 chunks of 16B; 128 threads x 2 chunks
  const int c0 = tid, c1 = tid + 128;
  const int r0 = c0 >> 3, ch0 = (c0 & 7) * 8;
  const int r1 = c1 >> 3, ch1 = (c1 & 7) * 8;
  const __bf16* ga0 = A + (size_t)(m0 + r0) * K + ch0;
  const __bf16* ga1 = A + (size_t)(m0 + r1) * K + ch1;

  // preload slab 0
  async_copy16(ga0, &Asl[0][r0][ch0]);
  async_copy16(ga1, &Asl[0][r1][ch1]);
  async_wait_all();
  __syncthreads();

  const int nslab = K >> 6;
  for (int ks = 0; ks < nslab; ++ks) {
    const int cur = ks & 1, nxt = cur ^ 1;
    const int K0 = ks << 6;

    if (ks + 1 < nslab) {               // prefetch next A slab (overlaps compute)
      async_copy16(ga0 + K0 + 64, &Asl[nxt][r0][ch0]);
      async_copy16(ga1 + K0 + 64, &Asl[nxt][r1][ch1]);
    }
#pragma unroll
    for (int t = 0; t < 4; ++t)         // warm B stream two slabs ahead
      __builtin_prefetch(bp[t] + K0 + 128, 0, 3);

    // all B loads for this slab (8 frags, 16 x b128)
    FragBF bf[2][4];
#pragma unroll
    for (int k2 = 0; k2 < 2; ++k2)
#pragma unroll
      for (int t = 0; t < 4; ++t) {
        bf[k2][t].q[0] = *reinterpret_cast<const uint4*>(bp[t] + K0 + k2 * 32);
        bf[k2][t].q[1] = *reinterpret_cast<const uint4*>(bp[t] + K0 + k2 * 32 + 8);
      }

    // A frags from LDS (4 frags, 8 x ds_read_b128)
    FragBF af[2][2];
#pragma unroll
    for (int mt = 0; mt < 2; ++mt)
#pragma unroll
      for (int k2 = 0; k2 < 2; ++k2) {
        const __bf16* lp = &Asl[cur][mt * 16 + lm][k2 * 32 + 16 * lg];
        af[mt][k2].q[0] = *reinterpret_cast<const uint4*>(lp);
        af[mt][k2].q[1] = *reinterpret_cast<const uint4*>(lp + 8);
      }

    // 16 WMMAs
#pragma unroll
    for (int k2 = 0; k2 < 2; ++k2)
#pragma unroll
      for (int mt = 0; mt < 2; ++mt)
#pragma unroll
        for (int t = 0; t < 4; ++t)
          acc[mt][t] = wmma_bf16(af[mt][k2], bf[k2][t], acc[mt][t]);

    async_wait_all();
    __syncthreads();
  }
}

// ---------------------------------------------------------------------------
// Generic GEMM: C = A @ B + bias (optional ReLU), writes f32 and/or bf16.
// grid: (Mtot/32, Ntot/256), block 128.
// ---------------------------------------------------------------------------
__global__ __launch_bounds__(128) void gemm_bf16(const __bf16* __restrict__ A,
                                                 const __bf16* __restrict__ Bt,
                                                 const float* __restrict__ bias,
                                                 float* __restrict__ Cf,
                                                 __bf16* __restrict__ Cb,
                                                 int Ntot, int K, int relu) {
  __shared__ __bf16 Asl[2][32][64] __attribute__((aligned(16)));
  const int tid = threadIdx.x;
  const int lane = tid & 31, wv = tid >> 5;
  const int lm = lane & 15, lg = lane >> 4;
  const int m0 = blockIdx.x * 32;
  const int n0 = blockIdx.y * 256 + wv * 64;

  v8f acc[2][4] = {};
  gemm_core_32x64(A, Bt, K, m0, n0, tid, Asl, acc);

#pragma unroll
  for (int mt = 0; mt < 2; ++mt) {
    const int mr = m0 + mt * 16 + 8 * lg;
#pragma unroll
    for (int t = 0; t < 4; ++t) {
      const int n = n0 + t * 16 + lm;
      const float bv = bias ? bias[n] : 0.0f;
#pragma unroll
      for (int r = 0; r < 8; ++r) {
        float v = acc[mt][t][r] + bv;
        if (relu) v = fmaxf(v, 0.0f);
        const size_t idx = (size_t)(mr + r) * Ntot + n;
        if (Cf) Cf[idx] = v;
        if (Cb) Cb[idx] = (__bf16)v;
      }
    }
  }
}

// ---------------------------------------------------------------------------
// QKV projection with fused RoPE + layout scatter.
// q,k -> [BH][S][64] (RoPE applied), v -> transposed [BH][64][S].
// grid: (ROWS_/32, 1536/256), block 128.
// ---------------------------------------------------------------------------
__global__ __launch_bounds__(128) void qkv_rope(const __bf16* __restrict__ A,
                                                const __bf16* __restrict__ Bt,
                                                const float* __restrict__ bias,
                                                __bf16* __restrict__ Q,
                                                __bf16* __restrict__ Kc,
                                                __bf16* __restrict__ Vt) {
  __shared__ __bf16 Asl[2][32][64] __attribute__((aligned(16)));
  const int tid = threadIdx.x;
  const int lane = tid & 31, wv = tid >> 5;
  const int lm = lane & 15, lg = lane >> 4;
  const int m0 = blockIdx.x * 32;
  const int n0 = blockIdx.y * 256 + wv * 64;

  v8f acc[2][4] = {};
  gemm_core_32x64(A, Bt, D_, m0, n0, tid, Asl, acc);

  const int sect = n0 / D_;            // 0=q, 1=k, 2=v
  const int hloc = (n0 % D_) / HD_;
  const int b    = m0 / S_;
  const int s0   = m0 % S_;
  const int bh   = b * H_ + hloc;

#pragma unroll
  for (int mt = 0; mt < 2; ++mt) {
    float c[4][8];
#pragma unroll
    for (int t = 0; t < 4; ++t) {
      const float bv = bias[n0 + t * 16 + lm];
#pragma unroll
      for (int r = 0; r < 8; ++r) c[t][r] = acc[mt][t][r] + bv;
    }
    const int mr = mt * 16 + 8 * lg;

    if (sect < 2) {
      __bf16* dst = (sect == 0) ? Q : Kc;
#pragma unroll
      for (int t2 = 0; t2 < 2; ++t2) {
        const int d = t2 * 16 + lm;                          // pairs with d+32
        const float invf = __expf((float)d * -0.2878231366f); // 10000^(-d/32)
#pragma unroll
        for (int r = 0; r < 8; ++r) {
          const int s = s0 + mr + r;
          float sn, cs;
          __sincosf((float)s * invf, &sn, &cs);
          const float x1 = c[t2][r], x2 = c[t2 + 2][r];
          const size_t base = ((size_t)bh * S_ + s) * HD_;
          dst[base + d]      = (__bf16)(x1 * cs - x2 * sn);
          dst[base + d + 32] = (__bf16)(x1 * sn + x2 * cs);
        }
      }
    } else {
#pragma unroll
      for (int t = 0; t < 4; ++t) {
        const int d = t * 16 + lm;
#pragma unroll
        for (int r = 0; r < 8; ++r) {
          const int s = s0 + mr + r;
          Vt[((size_t)bh * HD_ + d) * S_ + s] = (__bf16)c[t][r];
        }
      }
    }
  }
}

// ---------------------------------------------------------------------------
// Fused flash attention, software-pipelined. One wave owns 16 query rows.
// Key blocks of 32; V loads hoisted before scores; next K block prefetched
// into ping-pong register buffers (manual unroll x2 keeps parity static).
// grid: BH_*64/4 = 2048, block 128 (4 waves).
// ---------------------------------------------------------------------------
__global__ __launch_bounds__(128) void attn_fused(const __bf16* __restrict__ Qg,
                                                  const __bf16* __restrict__ Kg,
                                                  const __bf16* __restrict__ Vt,
                                                  __bf16* __restrict__ Og) {
  __shared__ __bf16 pb[4][16][32] __attribute__((aligned(16)));

  const int lane = threadIdx.x & 31;
  const int wv   = threadIdx.x >> 5;
  const int lm   = lane & 15;
  const int lg   = lane >> 4;
  const int wid  = blockIdx.x * 4 + wv;
  const int bh   = wid >> 6;
  const int mt   = wid & 63;
  const int b    = bh >> 3;
  const int h    = bh & 7;

  const __bf16* qp = Qg + ((size_t)bh * S_ + mt * 16 + lm) * HD_ + 8 * lg;
  FragBF qf0, qf1;
  qf0.q[0] = *reinterpret_cast<const uint4*>(qp);
  qf0.q[1] = *reinterpret_cast<const uint4*>(qp + 16);
  qf1.q[0] = *reinterpret_cast<const uint4*>(qp + 32);
  qf1.q[1] = *reinterpret_cast<const uint4*>(qp + 48);

  v8f o[4] = {};
  v8f mrow, lrow;
#pragma unroll
  for (int r = 0; r < 8; ++r) { mrow[r] = -1e30f; lrow[r] = 0.0f; }

  auto load_k = [&](FragBF (&kf)[2][2], int kn) {
#pragma unroll
    for (int t2 = 0; t2 < 2; ++t2) {
      const __bf16* kp = Kg + ((size_t)bh * S_ + kn + t2 * 16 + lm) * HD_ + 8 * lg;
      kf[t2][0].q[0] = *reinterpret_cast<const uint4*>(kp);
      kf[t2][0].q[1] = *reinterpret_cast<const uint4*>(kp + 16);
      kf[t2][1].q[0] = *reinterpret_cast<const uint4*>(kp + 32);
      kf[t2][1].q[1] = *reinterpret_cast<const uint4*>(kp + 48);
    }
  };

  auto step = [&](FragBF (&kcur)[2][2], FragBF (&knxt)[2][2], int kn, bool pref) {
    // V loads first: independent of P, latency hidden behind scores+softmax
    FragBF vf[4];
#pragma unroll
    for (int t = 0; t < 4; ++t) {
      const __bf16* vp = Vt + ((size_t)bh * HD_ + t * 16 + lm) * S_ + kn + 8 * lg;
      vf[t].q[0] = *reinterpret_cast<const uint4*>(vp);
      vf[t].q[1] = *reinterpret_cast<const uint4*>(vp + 16);
    }

    // scores from current K frags
    v8f st[2];
#pragma unroll
    for (int t2 = 0; t2 < 2; ++t2) {
      v8f s = {};
      s = wmma_bf16(qf0, kcur[t2][0], s);
      s = wmma_bf16(qf1, kcur[t2][1], s);
      st[t2] = s;
    }

    // prefetch next K block while softmax runs on the VALU
    if (pref) load_k(knxt, kn + 32);

    // online softmax
#pragma unroll
    for (int r = 0; r < 8; ++r) {
      st[0][r] *= 0.125f;
      st[1][r] *= 0.125f;
      float v = fmaxf(st[0][r], st[1][r]);
#pragma unroll
      for (int msk = 1; msk < 16; msk <<= 1) v = fmaxf(v, __shfl_xor(v, msk, 32));
      const float nm   = fmaxf(mrow[r], v);
      const float corr = __expf(mrow[r] - nm);
      st[0][r] = __expf(st[0][r] - nm);
      st[1][r] = __expf(st[1][r] - nm);
      float rs = st[0][r] + st[1][r];
#pragma unroll
      for (int msk = 1; msk < 16; msk <<= 1) rs += __shfl_xor(rs, msk, 32);
      lrow[r] = lrow[r] * corr + rs;
      mrow[r] = nm;
#pragma unroll
      for (int t = 0; t < 4; ++t) o[t][r] *= corr;
    }

    // P: C-layout -> A-fragment via per-wave LDS transpose
#pragma unroll
    for (int r = 0; r < 8; ++r) {
      pb[wv][8 * lg + r][lm]      = (__bf16)st[0][r];
      pb[wv][8 * lg + r][16 + lm] = (__bf16)st[1][r];
    }
    FragBF pf;
    pf.q[0] = *reinterpret_cast<const uint4*>(&pb[wv][lm][8 * lg]);
    pf.q[1] = *reinterpret_cast<const uint4*>(&pb[wv][lm][8 * lg + 16]);

    // O += P @ V
#pragma unroll
    for (int t = 0; t < 4; ++t) o[t] = wmma_bf16(pf, vf[t], o[t]);
  };

  FragBF ka[2][2], kb2[2][2];
  load_k(ka, 0);
  for (int jb = 0; jb < 32; jb += 2) {
    step(ka, kb2, jb * 32, true);
    step(kb2, ka, jb * 32 + 32, jb + 2 < 32);
  }

#pragma unroll
  for (int r = 0; r < 8; ++r) {
    const float inv = 1.0f / lrow[r];
    const size_t row = (size_t)b * S_ + mt * 16 + 8 * lg + r;
#pragma unroll
    for (int t = 0; t < 4; ++t)
      Og[row * D_ + h * HD_ + t * 16 + lm] = (__bf16)(o[t][r] * inv);
  }
}

// ---------------------------------------------------------------------------
// out = LayerNorm(a + b) * g + beta ; one block per row of 512.
// ---------------------------------------------------------------------------
__global__ __launch_bounds__(256) void add_ln(const float* __restrict__ A,
                                              const float* __restrict__ Br,
                                              const float* __restrict__ g,
                                              const float* __restrict__ beta,
                                              float* __restrict__ Of,
                                              __bf16* __restrict__ Ob) {
  const int row = blockIdx.x;
  const int t   = threadIdx.x;
  const float* pa = A  + (size_t)row * D_;
  const float* pr = Br + (size_t)row * D_;
  const float x0 = pa[t] + pr[t];
  const float x1 = pa[t + 256] + pr[t + 256];
  float s = x0 + x1, q = x0 * x0 + x1 * x1;
#pragma unroll
  for (int m = 1; m < 32; m <<= 1) { s += __shfl_xor(s, m, 32); q += __shfl_xor(q, m, 32); }
  __shared__ float ss[8], sq[8];
  if ((t & 31) == 0) { ss[t >> 5] = s; sq[t >> 5] = q; }
  __syncthreads();
  if (t == 0) {
    float a = 0.f, qq = 0.f;
    for (int i = 0; i < 8; ++i) { a += ss[i]; qq += sq[i]; }
    ss[0] = a; sq[0] = qq;
  }
  __syncthreads();
  const float mu   = ss[0] * (1.0f / D_);
  const float var  = sq[0] * (1.0f / D_) - mu * mu;
  const float rinv = rsqrtf(var + 1e-5f);
  const float y0 = (x0 - mu) * rinv * g[t] + beta[t];
  const float y1 = (x1 - mu) * rinv * g[t + 256] + beta[t + 256];
  const size_t o = (size_t)row * D_;
  if (Of) { Of[o + t] = y0; Of[o + t + 256] = y1; }
  if (Ob) { Ob[o + t] = (__bf16)y0; Ob[o + t + 256] = (__bf16)y1; }
}

// ---------------------------------------------------------------------------
// conversions
// ---------------------------------------------------------------------------
__global__ void f2bf_kernel(const float* __restrict__ in, __bf16* __restrict__ out, size_t n) {
  const size_t i = (size_t)blockIdx.x * blockDim.x + threadIdx.x;
  if (i < n) out[i] = (__bf16)in[i];
}

__global__ void transpose_f2bf(const float* __restrict__ w, __bf16* __restrict__ wt, int K, int N) {
  const int n = blockIdx.x * 32 + (threadIdx.x & 31);
  const int k = blockIdx.y * 8 + (threadIdx.x >> 5);
  if (n < N && k < K) wt[(size_t)n * K + k] = (__bf16)w[(size_t)k * N + n];
}

// ---------------------------------------------------------------------------
extern "C" void kernel_launch(void* const* d_in, const int* in_sizes, int n_in,
                              void* d_out, int out_size, void* d_ws, size_t ws_size,
                              hipStream_t stream) {
  (void)in_sizes; (void)n_in; (void)out_size; (void)ws_size;
  const float* x     = (const float*)d_in[0];
  const float* w_in  = (const float*)d_in[1];
  const float* b_in  = (const float*)d_in[2];
  const float* w_qkv = (const float*)d_in[3];
  const float* b_qkv = (const float*)d_in[4];
  const float* w_out = (const float*)d_in[5];
  const float* b_out = (const float*)d_in[6];
  const float* w_f1  = (const float*)d_in[7];
  const float* b_f1  = (const float*)d_in[8];
  const float* w_f2  = (const float*)d_in[9];
  const float* b_f2  = (const float*)d_in[10];
  const float* g1    = (const float*)d_in[11];
  const float* bn1   = (const float*)d_in[12];
  const float* g2    = (const float*)d_in[13];
  const float* bn2   = (const float*)d_in[14];

  char* ws = (char*)d_ws;
  size_t off = 0;
  auto alloc = [&](size_t bytes) -> char* {
    char* p = ws + off;
    off += (bytes + 255) & ~(size_t)255;
    return p;
  };

  __bf16* xb   = (__bf16*)alloc((size_t)ROWS_ * CIN_ * 2);
  __bf16* wti  = (__bf16*)alloc((size_t)D_ * CIN_ * 2);
  __bf16* wtq  = (__bf16*)alloc((size_t)3 * D_ * D_ * 2);
  __bf16* wto  = (__bf16*)alloc((size_t)D_ * D_ * 2);
  __bf16* wt1  = (__bf16*)alloc((size_t)4 * D_ * D_ * 2);
  __bf16* wt2  = (__bf16*)alloc((size_t)D_ * 4 * D_ * 2);
  float*  hf   = (float*) alloc((size_t)ROWS_ * D_ * 4);
  __bf16* hb   = (__bf16*)alloc((size_t)ROWS_ * D_ * 2);
  __bf16* qb   = (__bf16*)alloc((size_t)BH_ * S_ * HD_ * 2);
  __bf16* kb   = (__bf16*)alloc((size_t)BH_ * S_ * HD_ * 2);
  __bf16* vtb  = (__bf16*)alloc((size_t)BH_ * HD_ * S_ * 2);
  __bf16* ob   = (__bf16*)alloc((size_t)ROWS_ * D_ * 2);
  float*  opf  = (float*) alloc((size_t)ROWS_ * D_ * 4);
  float*  h2f  = (float*) alloc((size_t)ROWS_ * D_ * 4);
  __bf16* h2b  = (__bf16*)alloc((size_t)ROWS_ * D_ * 2);
  __bf16* fmid = (__bf16*)alloc((size_t)ROWS_ * 4 * D_ * 2);
  float*  ffo  = (float*) alloc((size_t)ROWS_ * D_ * 4);

  {
    const size_t n = (size_t)ROWS_ * CIN_;
    f2bf_kernel<<<(unsigned)((n + 255) / 256), 256, 0, stream>>>(x, xb, n);
  }
  transpose_f2bf<<<dim3((D_ + 31) / 32, (CIN_ + 7) / 8), 256, 0, stream>>>(w_in, wti, CIN_, D_);
  transpose_f2bf<<<dim3((3 * D_ + 31) / 32, (D_ + 7) / 8), 256, 0, stream>>>(w_qkv, wtq, D_, 3 * D_);
  transpose_f2bf<<<dim3((D_ + 31) / 32, (D_ + 7) / 8), 256, 0, stream>>>(w_out, wto, D_, D_);
  transpose_f2bf<<<dim3((4 * D_ + 31) / 32, (D_ + 7) / 8), 256, 0, stream>>>(w_f1, wt1, D_, 4 * D_);
  transpose_f2bf<<<dim3((D_ + 31) / 32, (4 * D_ + 7) / 8), 256, 0, stream>>>(w_f2, wt2, 4 * D_, D_);

  // h = x @ w_in + b_in
  gemm_bf16<<<dim3(ROWS_ / 32, D_ / 256), 128, 0, stream>>>(xb, wti, b_in, hf, hb, D_, CIN_, 0);

  // qkv + RoPE + scatter
  qkv_rope<<<dim3(ROWS_ / 32, (3 * D_) / 256), 128, 0, stream>>>(hb, wtq, b_qkv, qb, kb, vtb);

  // fused flash attention
  attn_fused<<<dim3(BH_ * (S_ / 16) / 4), 128, 0, stream>>>(qb, kb, vtb, ob);

  // out projection
  gemm_bf16<<<dim3(ROWS_ / 32, D_ / 256), 128, 0, stream>>>(ob, wto, b_out, opf, (__bf16*)nullptr, D_, D_, 0);

  // h2 = LN(h + o)
  add_ln<<<ROWS_, 256, 0, stream>>>(hf, opf, g1, bn1, h2f, h2b);

  // FFN1 with ReLU
  gemm_bf16<<<dim3(ROWS_ / 32, (4 * D_) / 256), 128, 0, stream>>>(h2b, wt1, b_f1, (float*)nullptr, fmid, 4 * D_, D_, 1);

  // FFN2
  gemm_bf16<<<dim3(ROWS_ / 32, D_ / 256), 128, 0, stream>>>(fmid, wt2, b_f2, ffo, (__bf16*)nullptr, D_, 4 * D_, 0);

  // out = LN(h2 + ffn)
  add_ln<<<ROWS_, 256, 0, stream>>>(h2f, ffo, g2, bn2, (float*)d_out, (__bf16*)nullptr);
}
